// EncoderBlock_57930518888921
// MI455X (gfx1250) — matrix operand order repaired and checked by reading, not compiled, not run on hardware
//
#include <hip/hip_runtime.h>
#include <hip/hip_bf16.h>

// ---------------------------------------------------------------------------
// CDNA5 (gfx1250) wave32 WMMA bf16 transformer encoder block.
// ---------------------------------------------------------------------------

#define S_SEQ  2048
#define DMODEL 1024
#define NHEAD  16
#define DHEAD  64
#define DMLP   4096

typedef __attribute__((ext_vector_type(16))) __bf16 v16bf;
typedef __attribute__((ext_vector_type(8)))  float  v8f;

union Frag16 {
  v16bf v;
  uint4 q[2];
  unsigned short s[16];
};

__device__ __forceinline__ unsigned short f2bf(float f) {
  union { float f; unsigned u; } c; c.f = f;
  unsigned u = c.u;
  u += 0x7FFFu + ((u >> 16) & 1u);   // round-to-nearest-even
  return (unsigned short)(u >> 16);
}

__device__ __forceinline__ v8f wmma_bf16(const Frag16& a, const Frag16& b, v8f c) {
  // D = A(16x32 bf16) * B(32x16 bf16) + C(16x16 f32)
  return __builtin_amdgcn_wmma_f32_16x16x32_bf16(false, a.v, false, b.v,
                                                 (short)0, c, false, false);
}

// ---------------------------------------------------------------------------
// fp32 -> bf16 weight conversion (vectorized, n is multiple of 4)
// ---------------------------------------------------------------------------
__global__ __launch_bounds__(256) void cvt_bf16_kernel(
    const float* __restrict__ src, unsigned short* __restrict__ dst, int n4) {
  int i = blockIdx.x * 256 + threadIdx.x;
  if (i < n4) {
    float4 f = ((const float4*)src)[i];
    ushort4 o;
    o.x = f2bf(f.x); o.y = f2bf(f.y); o.z = f2bf(f.z); o.w = f2bf(f.w);
    ((ushort4*)dst)[i] = o;
  }
}

// ---------------------------------------------------------------------------
// LayerNorm (one row of 1024 per block) -> bf16 output
// ---------------------------------------------------------------------------
__global__ __launch_bounds__(256) void ln_bf16_kernel(
    const float* __restrict__ x, const float* __restrict__ g,
    const float* __restrict__ b, unsigned short* __restrict__ out) {
  const int row = blockIdx.x;
  const int t = threadIdx.x;
  const float* xr = x + (size_t)row * DMODEL;
  float v[4]; float s = 0.f, s2 = 0.f;
#pragma unroll
  for (int i = 0; i < 4; ++i) {
    float f = xr[t + 256 * i]; v[i] = f; s += f; s2 += f * f;
  }
#pragma unroll
  for (int off = 16; off > 0; off >>= 1) {
    s  += __shfl_xor(s,  off, 32);
    s2 += __shfl_xor(s2, off, 32);
  }
  __shared__ float rs[8], rq[8];
  const int wid = t >> 5, lid = t & 31;
  if (lid == 0) { rs[wid] = s; rq[wid] = s2; }
  __syncthreads();
  s = 0.f; s2 = 0.f;
#pragma unroll
  for (int i = 0; i < 8; ++i) { s += rs[i]; s2 += rq[i]; }
  const float mean = s * (1.0f / DMODEL);
  const float var  = s2 * (1.0f / DMODEL) - mean * mean;
  const float r    = rsqrtf(var + 1e-5f);
#pragma unroll
  for (int i = 0; i < 4; ++i) {
    int c = t + 256 * i;
    out[(size_t)row * DMODEL + c] = f2bf((v[i] - mean) * r * g[c] + b[c]);
  }
}

// ---------------------------------------------------------------------------
// bf16 WMMA GEMM: C[M,N] = A[M,K] * B[K,N] (+bias, epilogue variants)
//   EPI 0: out = bf16(acc + bias)
//   EPI 1: out = f32 (acc + bias + residual)
//   EPI 2: out = bf16(gelu_exact(acc + bias))
// Block tile 128x128, BK=32, 8 waves -> each wave 32x64 (2x4 WMMA tiles).
// ---------------------------------------------------------------------------
template <int EPI>
__global__ __launch_bounds__(256) void gemm_bf16_kernel(
    const unsigned short* __restrict__ A, const unsigned short* __restrict__ B,
    const float* __restrict__ bias, const float* __restrict__ res,
    float* __restrict__ outF, unsigned short* __restrict__ outH,
    int M, int N, int K) {
  __shared__ unsigned short sA[128 * 40];  // [m][k], padded
  __shared__ unsigned short sB[128 * 40];  // transposed: [n][k], padded
  const int t = threadIdx.x;
  const int lane = t & 31, wave = t >> 5;
  const int hf = lane >> 4, ln = lane & 15;
  const int wm = wave >> 1, wn = wave & 1;
  const int m0 = blockIdx.y * 128, n0 = blockIdx.x * 128;

  v8f acc[2][4];
#pragma unroll
  for (int i = 0; i < 2; ++i)
#pragma unroll
    for (int j = 0; j < 4; ++j)
#pragma unroll
      for (int e = 0; e < 8; ++e) acc[i][j][e] = 0.f;

  for (int kb = 0; kb < K; kb += 32) {
    __syncthreads();
    // Stage A: 128x32 bf16, coalesced 16B loads.
#pragma unroll
    for (int i = 0; i < 2; ++i) {
      int idx = i * 256 + t;                  // 512 x uint4
      int r = idx >> 2, c8 = (idx & 3) * 8;
      uint4 d = *(const uint4*)(A + (size_t)(m0 + r) * K + kb + c8);
      *(uint4*)&sA[r * 40 + c8] = d;
    }
    // Stage B transposed: source rows [kb..kb+31], cols [n0..n0+127].
#pragma unroll
    for (int i = 0; i < 2; ++i) {
      int idx = i * 256 + t;                  // 512 x uint4
      int kr = idx >> 4, c8 = (idx & 15) * 8;
      uint4 d = *(const uint4*)(B + (size_t)(kb + kr) * N + n0 + c8);
      const unsigned short* ds = (const unsigned short*)&d;
#pragma unroll
      for (int j = 0; j < 8; ++j) sB[(c8 + j) * 40 + kr] = ds[j];
    }
    __syncthreads();

    Frag16 af[2], bf[4];
#pragma unroll
    for (int mt = 0; mt < 2; ++mt) {
      int m = wm * 32 + mt * 16 + ln;         // A frag: K halves interleaved
      af[mt].q[0] = *(const uint4*)&sA[m * 40 + hf * 8];
      af[mt].q[1] = *(const uint4*)&sA[m * 40 + 16 + hf * 8];
    }
#pragma unroll
    for (int nt = 0; nt < 4; ++nt) {
      int n = wn * 64 + nt * 16 + ln;         // B frag: K halves interleaved
      bf[nt].q[0] = *(const uint4*)&sB[n * 40 + hf * 8];
      bf[nt].q[1] = *(const uint4*)&sB[n * 40 + 16 + hf * 8];
    }
#pragma unroll
    for (int mt = 0; mt < 2; ++mt)
#pragma unroll
      for (int nt = 0; nt < 4; ++nt)
        acc[mt][nt] = wmma_bf16(af[mt], bf[nt], acc[mt][nt]);
  }

  // Epilogue: C layout m = j + 8*hf, n = ln within each 16x16 tile.
#pragma unroll
  for (int mt = 0; mt < 2; ++mt) {
#pragma unroll
    for (int nt = 0; nt < 4; ++nt) {
      int gc = n0 + wn * 64 + nt * 16 + ln;
      float bval = bias[gc];
#pragma unroll
      for (int j = 0; j < 8; ++j) {
        int gr = m0 + wm * 32 + mt * 16 + j + 8 * hf;
        float vv = acc[mt][nt][j] + bval;
        if (EPI == 0) {
          outH[(size_t)gr * N + gc] = f2bf(vv);
        } else if (EPI == 1) {
          outF[(size_t)gr * N + gc] = vv + res[(size_t)gr * N + gc];
        } else {
          float gl = 0.5f * vv * (1.0f + erff(vv * 0.70710678118654752f));
          outH[(size_t)gr * N + gc] = f2bf(gl);
        }
      }
    }
  }
}

// ---------------------------------------------------------------------------
// Fused flash-style attention. Grid: 16 heads * 16 query-blocks = 256 blocks.
// Block = 8 waves; each wave owns a 16-row query tile, streams keys in
// chunks of 32 with online softmax. K/V tiles staged in LDS cooperatively.
// Output written directly as the contiguous [H,S,DK] bf16 buffer (which IS
// the reference's o.reshape(S, D)).
// ---------------------------------------------------------------------------
__global__ __launch_bounds__(256) void attn_kernel(
    const unsigned short* __restrict__ q, const unsigned short* __restrict__ k,
    const unsigned short* __restrict__ v, unsigned short* __restrict__ o) {
  __shared__ unsigned short sK[32 * 72];       // [key][dim], padded
  __shared__ unsigned short sV[64 * 40];       // transposed: [dim][key]
  __shared__ unsigned short sP[8 * 16 * 40];   // per-wave P staging [m][key]
  const int t = threadIdx.x, lane = t & 31, wave = t >> 5;
  const int hf = lane >> 4, ln = lane & 15;
  const int head = blockIdx.x >> 4;
  const int q0 = (blockIdx.x & 15) * 128 + wave * 16;

  // Load Q A-fragments once (16 rows x 64 dims -> 2 frags of K=32).
  Frag16 qa[2];
  {
    const unsigned short* qp = q + (size_t)(q0 + ln) * DMODEL + head * DHEAD;
#pragma unroll
    for (int f = 0; f < 2; ++f) {
      qa[f].q[0] = *(const uint4*)(qp + f * 32 + hf * 8);
      qa[f].q[1] = *(const uint4*)(qp + f * 32 + 16 + hf * 8);
    }
  }

  v8f acc[4];
  float mst[8], lst[8];
#pragma unroll
  for (int nt = 0; nt < 4; ++nt)
#pragma unroll
    for (int j = 0; j < 8; ++j) acc[nt][j] = 0.f;
#pragma unroll
  for (int j = 0; j < 8; ++j) { mst[j] = -1e30f; lst[j] = 0.f; }

  unsigned short* pw = &sP[wave * 16 * 40];

  for (int kb = 0; kb < S_SEQ; kb += 32) {
    __syncthreads();
    {  // cooperative staging: 32 keys x 64 dims for this head
      int key = t >> 3, c8 = (t & 7) * 8;
      uint4 dk_ = *(const uint4*)(k + (size_t)(kb + key) * DMODEL + head * DHEAD + c8);
      *(uint4*)&sK[key * 72 + c8] = dk_;
      uint4 dv_ = *(const uint4*)(v + (size_t)(kb + key) * DMODEL + head * DHEAD + c8);
      const unsigned short* ds = (const unsigned short*)&dv_;
#pragma unroll
      for (int j = 0; j < 8; ++j) sV[(c8 + j) * 40 + key] = ds[j];
    }
    __syncthreads();

    // scores: S16x32 = Q(16x64) * K_chunk^T(64x32), two 16x16 tiles
    v8f sc[2];
#pragma unroll
    for (int kt = 0; kt < 2; ++kt) {
      Frag16 b0, b1;
      int key = kt * 16 + ln;
      b0.q[0] = *(const uint4*)&sK[key * 72 + hf * 8];
      b0.q[1] = *(const uint4*)&sK[key * 72 + 16 + hf * 8];
      b1.q[0] = *(const uint4*)&sK[key * 72 + 32 + hf * 8];
      b1.q[1] = *(const uint4*)&sK[key * 72 + 32 + 16 + hf * 8];
      v8f z;
#pragma unroll
      for (int e = 0; e < 8; ++e) z[e] = 0.f;
      z = wmma_bf16(qa[0], b0, z);
      z = wmma_bf16(qa[1], b1, z);
      sc[kt] = z;
    }

    // online softmax (rows live in one 16-lane half at fixed vgpr index)
    float corr[8];
#pragma unroll
    for (int j = 0; j < 8; ++j) {
      float s0 = sc[0][j] * 0.125f;          // 1/sqrt(64)
      float s1 = sc[1][j] * 0.125f;
      float rmax = fmaxf(s0, s1);
#pragma unroll
      for (int off = 8; off > 0; off >>= 1)
        rmax = fmaxf(rmax, __shfl_xor(rmax, off, 32));
      float mnew = fmaxf(mst[j], rmax);
      float cr = __expf(mst[j] - mnew);
      mst[j] = mnew;
      float p0 = __expf(s0 - mnew), p1 = __expf(s1 - mnew);
      float rsum = p0 + p1;
#pragma unroll
      for (int off = 8; off > 0; off >>= 1)
        rsum += __shfl_xor(rsum, off, 32);
      lst[j] = lst[j] * cr + rsum;
      corr[j] = cr;
      int m = j + 8 * hf;
      pw[m * 40 + ln]      = f2bf(p0);
      pw[m * 40 + 16 + ln] = f2bf(p1);
    }
#pragma unroll
    for (int nt = 0; nt < 4; ++nt)
#pragma unroll
      for (int j = 0; j < 8; ++j) acc[nt][j] *= corr[j];

    // cross-lane LDS hazard: wait for P-tile writes before fragment reads
    asm volatile("s_wait_dscnt 0" ::: "memory");

    Frag16 pa;
    pa.q[0] = *(const uint4*)&pw[ln * 40 + hf * 8];
    pa.q[1] = *(const uint4*)&pw[ln * 40 + 16 + hf * 8];
#pragma unroll
    for (int nt = 0; nt < 4; ++nt) {
      Frag16 vb;
      int dim = nt * 16 + ln;
      vb.q[0] = *(const uint4*)&sV[dim * 40 + hf * 8];
      vb.q[1] = *(const uint4*)&sV[dim * 40 + 16 + hf * 8];
      acc[nt] = wmma_bf16(pa, vb, acc[nt]);
    }
  }

  float inv[8];
#pragma unroll
  for (int j = 0; j < 8; ++j) inv[j] = 1.0f / lst[j];
#pragma unroll
  for (int nt = 0; nt < 4; ++nt)
#pragma unroll
    for (int j = 0; j < 8; ++j) {
      int row = q0 + j + 8 * hf;
      int dim = nt * 16 + ln;
      o[((size_t)head * S_SEQ + row) * DHEAD + dim] = f2bf(acc[nt][j] * inv[j]);
    }
}

// ---------------------------------------------------------------------------
// Host-side orchestration
// ---------------------------------------------------------------------------
extern "C" void kernel_launch(void* const* d_in, const int* in_sizes, int n_in,
                              void* d_out, int out_size, void* d_ws, size_t ws_size,
                              hipStream_t stream) {
  (void)in_sizes; (void)n_in; (void)out_size; (void)ws_size;
  const float* x    = (const float*)d_in[0];
  const float* wq   = (const float*)d_in[1];
  const float* bq   = (const float*)d_in[2];
  const float* wk   = (const float*)d_in[3];
  const float* bk   = (const float*)d_in[4];
  const float* wv   = (const float*)d_in[5];
  const float* bv   = (const float*)d_in[6];
  const float* wo   = (const float*)d_in[7];
  const float* bo   = (const float*)d_in[8];
  const float* w1   = (const float*)d_in[9];
  const float* b1   = (const float*)d_in[10];
  const float* w2   = (const float*)d_in[11];
  const float* b2   = (const float*)d_in[12];
  const float* ln1g = (const float*)d_in[13];
  const float* ln1b = (const float*)d_in[14];
  const float* ln2g = (const float*)d_in[15];
  const float* ln2b = (const float*)d_in[16];
  float* out = (float*)d_out;

  const size_t SD = (size_t)S_SEQ * DMODEL;       // 2M
  const size_t SF = (size_t)S_SEQ * DMLP;         // 8M
  const size_t WW = (size_t)DMODEL * DMODEL;      // 1M
  const size_t W1 = (size_t)DMODEL * DMLP;        // 4M

  char* p = (char*)d_ws;
  float* x1           = (float*)p;          p += SD * 4;  // fp32 residual
  unsigned short* hB  = (unsigned short*)p; p += SD * 2;  // ln1(x)
  unsigned short* qB  = (unsigned short*)p; p += SD * 2;
  unsigned short* kB  = (unsigned short*)p; p += SD * 2;
  unsigned short* vB  = (unsigned short*)p; p += SD * 2;
  unsigned short* oB  = (unsigned short*)p; p += SD * 2;  // attn out [H,S,DK]
  unsigned short* h2B = (unsigned short*)p; p += SD * 2;  // ln2(x1)
  unsigned short* fB  = (unsigned short*)p; p += SF * 2;  // gelu(ffn1)
  unsigned short* wqB = (unsigned short*)p; p += WW * 2;
  unsigned short* wkB = (unsigned short*)p; p += WW * 2;
  unsigned short* wvB = (unsigned short*)p; p += WW * 2;
  unsigned short* woB = (unsigned short*)p; p += WW * 2;
  unsigned short* w1B = (unsigned short*)p; p += W1 * 2;
  unsigned short* w2B = (unsigned short*)p; p += W1 * 2;

  // Weight conversion fp32 -> bf16
  cvt_bf16_kernel<<<(int)(WW / 4 / 256), 256, 0, stream>>>(wq, wqB, (int)(WW / 4));
  cvt_bf16_kernel<<<(int)(WW / 4 / 256), 256, 0, stream>>>(wk, wkB, (int)(WW / 4));
  cvt_bf16_kernel<<<(int)(WW / 4 / 256), 256, 0, stream>>>(wv, wvB, (int)(WW / 4));
  cvt_bf16_kernel<<<(int)(WW / 4 / 256), 256, 0, stream>>>(wo, woB, (int)(WW / 4));
  cvt_bf16_kernel<<<(int)(W1 / 4 / 256), 256, 0, stream>>>(w1, w1B, (int)(W1 / 4));
  cvt_bf16_kernel<<<(int)(W1 / 4 / 256), 256, 0, stream>>>(w2, w2B, (int)(W1 / 4));

  // LN1
  ln_bf16_kernel<<<S_SEQ, 256, 0, stream>>>(x, ln1g, ln1b, hB);

  // QKV projections
  dim3 gDD(DMODEL / 128, S_SEQ / 128);
  gemm_bf16_kernel<0><<<gDD, 256, 0, stream>>>(hB, wqB, bq, nullptr, nullptr, qB,
                                               S_SEQ, DMODEL, DMODEL);
  gemm_bf16_kernel<0><<<gDD, 256, 0, stream>>>(hB, wkB, bk, nullptr, nullptr, kB,
                                               S_SEQ, DMODEL, DMODEL);
  gemm_bf16_kernel<0><<<gDD, 256, 0, stream>>>(hB, wvB, bv, nullptr, nullptr, vB,
                                               S_SEQ, DMODEL, DMODEL);

  // Fused attention
  attn_kernel<<<NHEAD * (S_SEQ / 128), 256, 0, stream>>>(qB, kB, vB, oB);

  // Output projection + residual (x1 = x + o @ wo + bo)
  gemm_bf16_kernel<1><<<gDD, 256, 0, stream>>>(oB, woB, bo, x, x1, nullptr,
                                               S_SEQ, DMODEL, DMODEL);

  // LN2
  ln_bf16_kernel<<<S_SEQ, 256, 0, stream>>>(x1, ln2g, ln2b, h2B);

  // FFN1 + exact GELU
  dim3 gDM(DMLP / 128, S_SEQ / 128);
  gemm_bf16_kernel<2><<<gDM, 256, 0, stream>>>(h2B, w1B, b1, nullptr, nullptr, fB,
                                               S_SEQ, DMLP, DMODEL);

  // FFN2 + residual -> output
  gemm_bf16_kernel<1><<<gDD, 256, 0, stream>>>(fB, w2B, b2, x1, out, nullptr,
                                               S_SEQ, DMODEL, DMLP);
}
